// LinearRecurrentUnit_70317204570137
// MI455X (gfx1250) — matrix operand order repaired and checked by reading, not compiled, not run on hardware
//
#include <hip/hip_runtime.h>
#include <hip/hip_bf16.h>

typedef __attribute__((ext_vector_type(16))) __bf16 v16bf;
typedef __attribute__((ext_vector_type(8)))  __bf16 v8bf;
typedef __attribute__((ext_vector_type(4)))  __bf16 v4bf;
typedef __attribute__((ext_vector_type(8)))  float  v8f;
typedef __attribute__((ext_vector_type(4)))  unsigned int v4u;
typedef __attribute__((ext_vector_type(8)))  int v8i;
typedef __attribute__((ext_vector_type(4)))  int v4i;

// Problem sizes (fixed by the reference)
#define N_BATCH 8
#define T_LEN   4096
#define D_MODEL 512
#define DIM     512
#define M_ROWS  (N_BATCH * T_LEN)   // 32768
#define N_CHUNK 32
#define CHUNK_L 128                 // 32*128 = 4096 = T_LEN
#define LDSS    40                  // LDS bf16 row stride: 80B rows (64B data + 16B TDM pad)
#define BPLANE  (64 * LDSS)         // one staged B plane: 2560 bf16

// ---------------------------------------------------------------------------
// bf16 split helpers (bf16x3 f32 emulation on the WMMA pipe)
// ---------------------------------------------------------------------------
__device__ __forceinline__ void split2(float v, __bf16 &hi, __bf16 &lo) {
  hi = (__bf16)v;
  lo = (__bf16)(v - (float)hi);
}

__device__ __forceinline__ void split4(float4 v, v4bf &hi, v4bf &lo) {
  __bf16 h, l;
  split2(v.x, h, l); hi[0] = h; lo[0] = l;
  split2(v.y, h, l); hi[1] = h; lo[1] = l;
  split2(v.z, h, l); hi[2] = h; lo[2] = l;
  split2(v.w, h, l); hi[3] = h; lo[3] = l;
}

__device__ __forceinline__ v16bf join16(v8bf a, v8bf b) {
  v16bf r;
#pragma unroll
  for (int i = 0; i < 8; ++i) { r[i] = a[i]; r[i + 8] = b[i]; }
  return r;
}

// A fragment (16x32 bf16): per lane K = koff+0..7 then koff+16..23, koff = 8*(lane/16)
__device__ __forceinline__ v16bf load_frag_A(const __bf16* s, int row0, int lane) {
  const __bf16* p = s + (row0 + (lane & 15)) * LDSS + ((lane >> 4) << 3);
  return join16(*(const v8bf*)p, *(const v8bf*)(p + 16));
}

// B fragment (32x16 bf16): per lane K = koff+0..7 then koff+16..23, koff = 8*(lane/16); sT is [n][k]
__device__ __forceinline__ v16bf load_frag_B(const __bf16* s, int col0, int lane) {
  const __bf16* p = s + (col0 + (lane & 15)) * LDSS + ((lane >> 4) << 3);
  return join16(*(const v8bf*)p, *(const v8bf*)(p + 16));
}

__device__ __forceinline__ v8f wmma_bf16x3(const v16bf &ah, const v16bf &al,
                                           const v16bf &bh, const v16bf &bl, v8f c) {
  c = __builtin_amdgcn_wmma_f32_16x16x32_bf16(false, ah, false, bh, (short)0, c, false, false);
  c = __builtin_amdgcn_wmma_f32_16x16x32_bf16(false, al, false, bh, (short)0, c, false, false);
  c = __builtin_amdgcn_wmma_f32_16x16x32_bf16(false, ah, false, bl, (short)0, c, false, false);
  return c;
}

// ---------------------------------------------------------------------------
// TDM: move four contiguous 32(k) x 64(n) bf16 weight tiles (hi/lo x re/im
// planes, plane pitch 512*512 elems) into LDS in one 3-D descriptor.
// ISA 8.3/8.4/8.5: count=1, type=2, data_size=1(2B), pad 16B after every 64B
// row -> LDS pitch 80B (= LDSS bf16). workgroup_mask=0 (not in a cluster).
// This toolchain's builtin takes 6 args (clang-23 form): the 5th int32x8 is
// beyond the ISA's VADDR0-3 descriptor groups -> zero-filled.
// ---------------------------------------------------------------------------
__device__ __forceinline__ void tdm_load_b4(const __bf16* gsrc, __bf16* ldst) {
  const unsigned long long ga = (unsigned long long)(size_t)(const void*)gsrc;
  const unsigned int la = (unsigned int)(size_t)(void*)ldst;   // generic lo32 = LDS offset
  v4u g0;
  g0[0] = 1u;                                            // count=1 (user D#)
  g0[1] = la;                                            // lds_addr
  g0[2] = (unsigned int)(ga & 0xffffffffull);            // global_addr[31:0]
  g0[3] = (unsigned int)((ga >> 32) & 0x1ffffffull) | (2u << 30);  // addr[56:32] | type=2
  v8i g1;
  g1[0] = (int)((1u << 16)      // data_size = 2 bytes
              | (1u << 20)      // pad_enable
              | (3u << 22)      // pad_interval: 16 DWORDs (64B) between pads
              | (3u << 25));    // pad_amount: 4 DWORDs (16B)
  g1[1] = (int)(512u << 16);    // tensor_dim0 = 512 (k extent)
  g1[2] = (int)(512u << 16);    // tensor_dim0 hi | tensor_dim1 = 512 (n extent)
  g1[3] = (int)(32u << 16);     // tensor_dim1 hi | tile_dim0 = 32 (k)
  g1[4] = (int)(64u | (4u << 16)); // tile_dim1 = 64 (n) | tile_dim2 = 4 (planes)
  g1[5] = 512;                  // tensor_dim0_stride = 512 (row pitch, elems)
  g1[6] = 0;                    // dim0_stride hi | tensor_dim1_stride lo16 (262144 & 0xffff = 0)
  g1[7] = 4;                    // tensor_dim1_stride hi (262144 >> 16) : plane pitch
  v4i g2 = {4, 0, 0, 0};        // tensor_dim2 = 4 (planes); no dim3
  v4i g3 = {0, 0, 0, 0};
  v8i g4 = {0, 0, 0, 0, 0, 0, 0, 0};   // reserved trailing group (zero)
  __builtin_amdgcn_tensor_load_to_lds(g0, g1, g2, g3, g4, 0);
}

// ---------------------------------------------------------------------------
// prep_w: one-time weight conversion into transposed bf16 hi/lo planes.
//   BnT[n][d] = B[d][n] * e^gamma[n]   (re, im)
//   CT [d][n] = C[n][d]                (re, im negated)
// ---------------------------------------------------------------------------
__global__ __launch_bounds__(256) void lru_prep_w(
    const float* __restrict__ Bre, const float* __restrict__ Bim,
    const float* __restrict__ gamma_log,
    const float* __restrict__ Cre, const float* __restrict__ Cim,
    __bf16* __restrict__ Brh, __bf16* __restrict__ Brl,
    __bf16* __restrict__ Bih, __bf16* __restrict__ Bil,
    __bf16* __restrict__ Crh, __bf16* __restrict__ Crl,
    __bf16* __restrict__ Cih, __bf16* __restrict__ Cil) {
  const int idx = blockIdx.x * 256 + threadIdx.x;   // 262144
  const int c = idx & 511;
  const int r = idx >> 9;
  const size_t in  = (size_t)c * 512 + r;
  const size_t out = (size_t)r * 512 + c;
  const float g = __expf(gamma_log[r]);
  __bf16 h, l;
  split2(Bre[in] * g, h, l); Brh[out] = h; Brl[out] = l;
  split2(Bim[in] * g, h, l); Bih[out] = h; Bil[out] = l;
  split2( Cre[in],    h, l); Crh[out] = h; Crl[out] = l;
  split2(-Cim[in],    h, l); Cih[out] = h; Cil[out] = l;
}

// Stage one 128 x 32 f32 tile as bf16 hi/lo planes (split once per element).
__device__ __forceinline__ void stage_a_f32(const float* __restrict__ src, size_t row0,
                                            int kc, __bf16* sh, __bf16* sl, int tid) {
#pragma unroll
  for (int r = 0; r < 4; ++r) {
    const int e   = tid + r * 256;         // 1024 float4s
    const int row = e >> 3;
    const int c4  = (e & 7) << 2;
    const float4 v = *(const float4*)(src + (row0 + row) * D_MODEL + kc + c4);
    v4bf hi, lo;
    split4(v, hi, lo);
    *(v4bf*)&sh[row * LDSS + c4] = hi;
    *(v4bf*)&sl[row * LDSS + c4] = lo;
  }
}

// ---------------------------------------------------------------------------
// GEMM1: Bu_re = x @ (B_re e^g), Bu_im = x @ (B_im e^g)
// block tile 128x64, 8 waves of 32x32, K-step 32; B tiles via TDM
// ---------------------------------------------------------------------------
__global__ __launch_bounds__(256) void lru_gemm1(const float* __restrict__ x,
                                                 const __bf16* __restrict__ Wb,  // 4 contiguous planes
                                                 float* __restrict__ Hre,
                                                 float* __restrict__ Him) {
  __shared__ __align__(16) __bf16 sAh[128 * LDSS];
  __shared__ __align__(16) __bf16 sAl[128 * LDSS];
  __shared__ __align__(16) __bf16 sB[4 * BPLANE];   // hi_re, lo_re, hi_im, lo_im

  const int tid = threadIdx.x;
  const int m0  = blockIdx.x * 128;
  const int n0  = blockIdx.y * 64;
  const int lane = tid & 31;
  const int wid  = tid >> 5;
  const int wm   = (wid & 3) * 32;
  const int wn   = (wid >> 2) * 32;

  const v8f vz = {0.f, 0.f, 0.f, 0.f, 0.f, 0.f, 0.f, 0.f};
  v8f accr[2][2], acci[2][2];
#pragma unroll
  for (int i = 0; i < 2; ++i)
#pragma unroll
    for (int j = 0; j < 2; ++j) { accr[i][j] = vz; acci[i][j] = vz; }

  for (int kc = 0; kc < DIM; kc += 32) {
    if (wid == 0) tdm_load_b4(Wb + (size_t)n0 * 512 + kc, sB);
    stage_a_f32(x, (size_t)m0, kc, sAh, sAl, tid);   // overlaps with in-flight TDM
    if (wid == 0) __builtin_amdgcn_s_wait_tensorcnt(0);
    __syncthreads();

    v16bf ah[2], al[2];
#pragma unroll
    for (int tm = 0; tm < 2; ++tm) {
      ah[tm] = load_frag_A(sAh, wm + tm * 16, lane);
      al[tm] = load_frag_A(sAl, wm + tm * 16, lane);
    }
#pragma unroll
    for (int tn = 0; tn < 2; ++tn) {
      const v16bf brh = load_frag_B(sB + 0 * BPLANE, wn + tn * 16, lane);
      const v16bf brl = load_frag_B(sB + 1 * BPLANE, wn + tn * 16, lane);
      const v16bf bih = load_frag_B(sB + 2 * BPLANE, wn + tn * 16, lane);
      const v16bf bil = load_frag_B(sB + 3 * BPLANE, wn + tn * 16, lane);
#pragma unroll
      for (int tm = 0; tm < 2; ++tm) {
        accr[tm][tn] = wmma_bf16x3(ah[tm], al[tm], brh, brl, accr[tm][tn]);
        acci[tm][tn] = wmma_bf16x3(ah[tm], al[tm], bih, bil, acci[tm][tn]);
      }
    }
    __syncthreads();
  }

  // epilogue: D layout — VGPR r: M = r + 8*(lane/16), N = lane%16
#pragma unroll
  for (int tm = 0; tm < 2; ++tm)
#pragma unroll
    for (int tn = 0; tn < 2; ++tn)
#pragma unroll
      for (int r = 0; r < 8; ++r) {
        const int row = m0 + wm + tm * 16 + (lane >> 4) * 8 + r;
        const int col = n0 + wn + tn * 16 + (lane & 15);
        Hre[(size_t)row * DIM + col] = accr[tm][tn][r];
        Him[(size_t)row * DIM + col] = acci[tm][tn][r];
      }
}

// ---------------------------------------------------------------------------
// Scan phase A: in-place local scans of 32 chunks x 128 steps per (b, n)
// ---------------------------------------------------------------------------
__global__ __launch_bounds__(256) void lru_scan_a(const float* __restrict__ nu_log,
                                                  const float* __restrict__ theta_log,
                                                  float* __restrict__ Hre,
                                                  float* __restrict__ Him,
                                                  float* __restrict__ Sre,
                                                  float* __restrict__ Sim) {
  const int idx   = blockIdx.x * 256 + threadIdx.x;   // 131072 total
  const int n     = idx & (DIM - 1);
  const int chunk = (idx >> 9) & (N_CHUNK - 1);
  const int b     = idx >> 14;

  const float nu  = __expf(nu_log[n]);
  const float th  = __expf(theta_log[n]);
  const float mag = __expf(-nu);
  const float lr  = mag * cosf(th);
  const float li  = mag * sinf(th);

  size_t o = ((size_t)(b * T_LEN + chunk * CHUNK_L)) * DIM + n;
  float hre = 0.f, him = 0.f;
  for (int i = 0; i < CHUNK_L; ++i, o += DIM) {
    const float bre = Hre[o];
    const float bim = Him[o];
    const float nr = lr * hre - li * him + bre;
    const float ni = lr * him + li * hre + bim;
    hre = nr; him = ni;
    Hre[o] = hre; Him[o] = him;
  }
  const size_t so = (size_t)(b * N_CHUNK + chunk) * DIM + n;
  Sre[so] = hre; Sim[so] = him;
}

// ---------------------------------------------------------------------------
// Scan phase B: per (b,n) exclusive scan of chunk carries with Lambda^128
// (closed-form: Lambda^L = exp(-L e^nu) * cis(L e^theta)); c_0 = h0
// ---------------------------------------------------------------------------
__global__ __launch_bounds__(256) void lru_scan_b(const float* __restrict__ nu_log,
                                                  const float* __restrict__ theta_log,
                                                  const float* __restrict__ h0,
                                                  float* __restrict__ Sre,
                                                  float* __restrict__ Sim) {
  const int idx = blockIdx.x * 256 + threadIdx.x;     // 4096 total
  const int n   = idx & (DIM - 1);
  const int b   = idx >> 9;

  const float nu  = __expf(nu_log[n]);
  const float th  = __expf(theta_log[n]);
  const float mag = __expf(-(float)CHUNK_L * nu);
  const float ang = (float)CHUNK_L * th;
  const float Lr  = mag * cosf(ang);
  const float Li  = mag * sinf(ang);

  float cre = h0[b * DIM + n];   // h0 is real f32 cast to complex in ref
  float cim = 0.f;
  for (int j = 0; j < N_CHUNK; ++j) {
    const size_t o = (size_t)(b * N_CHUNK + j) * DIM + n;
    const float sre = Sre[o];
    const float sim = Sim[o];
    Sre[o] = cre; Sim[o] = cim;                       // exclusive carry
    const float tr = Lr * cre - Li * cim + sre;
    cim = Lr * cim + Li * cre + sim;
    cre = tr;
  }
}

// ---------------------------------------------------------------------------
// Scan phase C: H_t += Lambda^{i+1} * carry ; also emits new_h (t = T-1)
// ---------------------------------------------------------------------------
__global__ __launch_bounds__(256) void lru_scan_c(const float* __restrict__ nu_log,
                                                  const float* __restrict__ theta_log,
                                                  const float* __restrict__ Sre,
                                                  const float* __restrict__ Sim,
                                                  float* __restrict__ Hre,
                                                  float* __restrict__ Him,
                                                  float* __restrict__ new_h) {
  const int idx   = blockIdx.x * 256 + threadIdx.x;
  const int n     = idx & (DIM - 1);
  const int chunk = (idx >> 9) & (N_CHUNK - 1);
  const int b     = idx >> 14;

  const float nu  = __expf(nu_log[n]);
  const float th  = __expf(theta_log[n]);
  const float mag = __expf(-nu);
  const float lr  = mag * cosf(th);
  const float li  = mag * sinf(th);

  const size_t so = (size_t)(b * N_CHUNK + chunk) * DIM + n;
  const float cre = Sre[so];
  const float cim = Sim[so];

  float pr = lr, pi = li;        // Lambda^{i+1}
  size_t o = ((size_t)(b * T_LEN + chunk * CHUNK_L)) * DIM + n;
  float hre = 0.f, him = 0.f;
  for (int i = 0; i < CHUNK_L; ++i, o += DIM) {
    hre = Hre[o] + (pr * cre - pi * cim);
    him = Him[o] + (pr * cim + pi * cre);
    Hre[o] = hre; Him[o] = him;
    const float npr = pr * lr - pi * li;
    pi = pr * li + pi * lr;
    pr = npr;
  }
  if (chunk == N_CHUNK - 1) {    // last timestep -> new_h (complex64 interleaved)
    new_h[(size_t)(b * DIM + n) * 2 + 0] = hre;
    new_h[(size_t)(b * DIM + n) * 2 + 1] = him;
  }
}

// ---------------------------------------------------------------------------
// GEMM2: y = Hre @ Cre - Him @ Cim + D * x   (Cim negated in prep_w)
// ---------------------------------------------------------------------------
__global__ __launch_bounds__(256) void lru_gemm2(const float* __restrict__ Hre,
                                                 const float* __restrict__ Him,
                                                 const __bf16* __restrict__ Wc,  // 4 contiguous planes
                                                 const float* __restrict__ Dp,
                                                 const float* __restrict__ x,
                                                 float* __restrict__ y) {
  __shared__ __align__(16) __bf16 sArh[128 * LDSS];
  __shared__ __align__(16) __bf16 sArl[128 * LDSS];
  __shared__ __align__(16) __bf16 sAih[128 * LDSS];
  __shared__ __align__(16) __bf16 sAil[128 * LDSS];
  __shared__ __align__(16) __bf16 sB[4 * BPLANE];
  __shared__ float sD[64];

  const int tid = threadIdx.x;
  const int m0  = blockIdx.x * 128;
  const int n0  = blockIdx.y * 64;
  if (tid < 64) sD[tid] = Dp[n0 + tid];

  const int lane = tid & 31;
  const int wid  = tid >> 5;
  const int wm   = (wid & 3) * 32;
  const int wn   = (wid >> 2) * 32;

  const v8f vz = {0.f, 0.f, 0.f, 0.f, 0.f, 0.f, 0.f, 0.f};
  v8f acc[2][2];
#pragma unroll
  for (int i = 0; i < 2; ++i)
#pragma unroll
    for (int j = 0; j < 2; ++j) acc[i][j] = vz;

  __syncthreads();

  for (int kc = 0; kc < DIM; kc += 32) {
    if (wid == 0) tdm_load_b4(Wc + (size_t)n0 * 512 + kc, sB);
    stage_a_f32(Hre, (size_t)m0, kc, sArh, sArl, tid);
    stage_a_f32(Him, (size_t)m0, kc, sAih, sAil, tid);
    if (wid == 0) __builtin_amdgcn_s_wait_tensorcnt(0);
    __syncthreads();

    v16bf arh[2], arl[2], aih[2], ail[2];
#pragma unroll
    for (int tm = 0; tm < 2; ++tm) {
      arh[tm] = load_frag_A(sArh, wm + tm * 16, lane);
      arl[tm] = load_frag_A(sArl, wm + tm * 16, lane);
      aih[tm] = load_frag_A(sAih, wm + tm * 16, lane);
      ail[tm] = load_frag_A(sAil, wm + tm * 16, lane);
    }
#pragma unroll
    for (int tn = 0; tn < 2; ++tn) {
      const v16bf brh = load_frag_B(sB + 0 * BPLANE, wn + tn * 16, lane);
      const v16bf brl = load_frag_B(sB + 1 * BPLANE, wn + tn * 16, lane);
      const v16bf bih = load_frag_B(sB + 2 * BPLANE, wn + tn * 16, lane);
      const v16bf bil = load_frag_B(sB + 3 * BPLANE, wn + tn * 16, lane);
#pragma unroll
      for (int tm = 0; tm < 2; ++tm) {
        acc[tm][tn] = wmma_bf16x3(arh[tm], arl[tm], brh, brl, acc[tm][tn]);
        acc[tm][tn] = wmma_bf16x3(aih[tm], ail[tm], bih, bil, acc[tm][tn]);
      }
    }
    __syncthreads();
  }

#pragma unroll
  for (int tm = 0; tm < 2; ++tm)
#pragma unroll
    for (int tn = 0; tn < 2; ++tn)
#pragma unroll
      for (int r = 0; r < 8; ++r) {
        const int row = m0 + wm + tm * 16 + (lane >> 4) * 8 + r;
        const int col = n0 + wn + tn * 16 + (lane & 15);
        const size_t go = (size_t)row * D_MODEL + col;
        y[go] = acc[tm][tn][r] + sD[col - n0] * x[go];
      }
}

// ---------------------------------------------------------------------------
extern "C" void kernel_launch(void* const* d_in, const int* in_sizes, int n_in,
                              void* d_out, int out_size, void* d_ws, size_t ws_size,
                              hipStream_t stream) {
  const float* x         = (const float*)d_in[0];
  const float* h0        = (const float*)d_in[1];
  const float* nu_log    = (const float*)d_in[2];
  const float* theta_log = (const float*)d_in[3];
  const float* B_re      = (const float*)d_in[4];
  const float* B_im      = (const float*)d_in[5];
  const float* C_re      = (const float*)d_in[6];
  const float* C_im      = (const float*)d_in[7];
  const float* D_p       = (const float*)d_in[8];
  const float* gamma_log = (const float*)d_in[9];

  float* out = (float*)d_out;                    // y: 32768*512 f32, then new_h complex64
  float* Hre = (float*)d_ws;                     // 64 MB (Bu -> H in place)
  float* Him = Hre + (size_t)M_ROWS * DIM;       // 64 MB
  float* Sre = Him + (size_t)M_ROWS * DIM;       // 0.5 MB carries
  float* Sim = Sre + (size_t)N_BATCH * N_CHUNK * DIM;

  __bf16* W   = (__bf16*)(Sim + (size_t)N_BATCH * N_CHUNK * DIM);
  const size_t WP = (size_t)DIM * D_MODEL;       // 262144 elems per plane
  __bf16 *Brh = W,          *Brl = W + WP,     *Bih = W + 2 * WP, *Bil = W + 3 * WP;
  __bf16 *Crh = W + 4 * WP, *Crl = W + 5 * WP, *Cih = W + 6 * WP, *Cil = W + 7 * WP;

  float* y     = out;
  float* new_h = out + (size_t)M_ROWS * D_MODEL; // 8192 floats (interleaved complex)

  dim3 gemm_grid(M_ROWS / 128, DIM / 64);        // 256 x 8

  lru_prep_w<<<(DIM * D_MODEL) / 256, 256, 0, stream>>>(B_re, B_im, gamma_log, C_re, C_im,
                                                        Brh, Brl, Bih, Bil, Crh, Crl, Cih, Cil);
  lru_gemm1<<<gemm_grid, 256, 0, stream>>>(x, Brh, Hre, Him);
  lru_scan_a<<<(N_BATCH * N_CHUNK * DIM) / 256, 256, 0, stream>>>(nu_log, theta_log, Hre, Him, Sre, Sim);
  lru_scan_b<<<(N_BATCH * DIM) / 256, 256, 0, stream>>>(nu_log, theta_log, h0, Sre, Sim);
  lru_scan_c<<<(N_BATCH * N_CHUNK * DIM) / 256, 256, 0, stream>>>(nu_log, theta_log, Sre, Sim, Hre, Him, new_h);
  lru_gemm2<<<gemm_grid, 256, 0, stream>>>(Hre, Him, Crh, D_p, x, y);
}
